// Arch23GraphEncoder_8546984919436
// MI455X (gfx1250) — matrix operand order repaired and checked by reading, not compiled, not run on hardware
//
#include <hip/hip_runtime.h>
#include <hip/hip_bf16.h>
#include <stdint.h>

// ---------------- problem constants ----------------
#define B_GRAPHS 32
#define NPG      128
#define N_TOTAL  4096
#define M_SUB    4
#define K_SUB    16
#define S_SUB    16384          // subgraphs (tokens)
#define FLAT_N   262144         // flat subgraph nodes
#define H_DIM    128
#define NHEAD    4
#define DHEAD    32
#define RW_STEPS 16
#define NL_GNN   4
#define NL_RO    2
#define E_GLOB   32768
#define E_INTRA  524288
#define FFN_DIM  512

typedef __attribute__((ext_vector_type(16))) _Float16 v16h;
typedef __attribute__((ext_vector_type(8)))  float    v8f;

#if defined(__has_builtin)
#if __has_builtin(__builtin_amdgcn_global_load_async_to_lds_b128)
#define USE_ASYNC_B 1
// exact param types per clang diagnostics:
//   arg0: AS(1) pointer to 16-byte int vector
//   arg1: AS(3) pointer to 16-byte int vector
typedef __attribute__((__vector_size__(4 * sizeof(int)))) int v4i_async;
typedef __attribute__((address_space(1))) v4i_async* g_v4i_ptr;
typedef __attribute__((address_space(3))) v4i_async* l_v4i_ptr;
#endif
#endif

// =====================================================================
// Generic WMMA GEMM:  C[M,N] (= or +=) act( A[M,K](f32) @ B[K,N](f16) + bias )
// Requires M%128==0, N%128==0, K%32==0 (true for every call here).
// Block = 256 threads (8 waves), tile = 128x128, double-buffered LDS:
// chunk k+1 is staged (A: f32->f16 VALU convert, B: async copy when
// available) while chunk k feeds v_wmma. In-place A==C is safe: each
// block reads all of its 128 A-rows (staging) before any store.
// =====================================================================
__global__ void __launch_bounds__(256)
gemm_wmma(const float* A, const _Float16* __restrict__ Bw,
          const float* __restrict__ bias, float* C,
          int M, int N, int K, int act, int addFlag)
{
    __shared__ _Float16 As[2][128 * 32];   // [buf][row][k]  2 x 8 KB
    __shared__ _Float16 Bs[2][32 * 128];   // [buf][k][col]  2 x 8 KB

    const int mB   = blockIdx.x;
    const int nB   = blockIdx.y;
    const int t    = threadIdx.x;
    const int w    = t >> 5;            // wave id 0..7
    const int lane = t & 31;
    const int g    = lane >> 4;         // lane half 0/1
    const int lr   = lane & 15;

    v8f acc[8];
#pragma unroll
    for (int i = 0; i < 8; ++i)
#pragma unroll
        for (int j = 0; j < 8; ++j) acc[i][j] = 0.0f;

    const int aRow = t >> 1;            // 0..127
    const int aCol = (t & 1) * 16;      // 0/16
    const int bRow = t >> 3;            // 0..31
    const int bCol = (t & 7) * 16;      // 0..112

    auto stageA = [&](int buf, int kc) {
        const float* src = A + (size_t)(mB * 128 + aRow) * K + kc + aCol;
        _Float16* dst = &As[buf][aRow * 32 + aCol];
#pragma unroll
        for (int i = 0; i < 16; ++i) dst[i] = (_Float16)src[i];
    };
    auto stageB = [&](int buf, int kc) {
        const _Float16* src = Bw + (size_t)(kc + bRow) * N + nB * 128 + bCol;
        _Float16* dst = &Bs[buf][bRow * 128 + bCol];
#ifdef USE_ASYNC_B
        // 16 halves = 32 bytes per thread -> two async b128 transfers
        __builtin_amdgcn_global_load_async_to_lds_b128(
            (g_v4i_ptr)src, (l_v4i_ptr)dst, 0, 0);
        __builtin_amdgcn_global_load_async_to_lds_b128(
            (g_v4i_ptr)(src + 8), (l_v4i_ptr)(dst + 8), 0, 0);
#else
#pragma unroll
        for (int i = 0; i < 16; ++i) dst[i] = src[i];
#endif
    };

    // prologue: stage first chunk
    stageA(0, 0);
    stageB(0, 0);
#ifdef USE_ASYNC_B
    asm volatile("s_wait_asynccnt 0x0" ::: "memory");
#endif
    __syncthreads();

    int buf = 0;
    for (int kc = 0; kc < K; kc += 32, buf ^= 1) {
        const int kn = kc + 32;
        if (kn < K) {
            if (kn + 32 < K)    // pull A lines for the chunk after next
                __builtin_prefetch(A + (size_t)(mB * 128 + aRow) * K + kn + 32 + aCol, 0, 1);
            stageA(buf ^ 1, kn);
            stageB(buf ^ 1, kn);
        }

        // A fragment: lane L holds row M = w*16 + (L%16); per ISA table the
        // 8 VGPRs hold K-pairs {2r,2r+1}(+8 for upper lane half, +16 for r>=4)
        v16h af;
        const _Float16* arow = &As[buf][(w * 16 + lr) * 32];
#pragma unroll
        for (int r = 0; r < 8; ++r) {
            int kb = ((r < 4) ? (2 * r) : (2 * r + 8)) + 8 * g;
            af[2 * r]     = arow[kb];
            af[2 * r + 1] = arow[kb + 1];
        }

#pragma unroll
        for (int nt = 0; nt < 8; ++nt) {
            // B fragment: lane L = N column (L%16), 16 slots = K rows in table order
            v16h bf;
            const _Float16* brow = &Bs[buf][nt * 16 + lr];
#pragma unroll
            for (int e = 0; e < 16; ++e) bf[e] = brow[(((e < 8) ? e : (e + 8)) + 8 * g) * 128];
            acc[nt] = __builtin_amdgcn_wmma_f32_16x16x32_f16(
                false, af, false, bf, (short)0, acc[nt], false, false);
        }

#ifdef USE_ASYNC_B
        asm volatile("s_wait_asynccnt 0x0" ::: "memory");
#endif
        __syncthreads();
    }

    // C/D layout: lane L -> col = nt*16 + (L%16); VGPR vr -> row vr + 8*(L/16)
    const int rowBase = mB * 128 + w * 16 + 8 * g;
#pragma unroll
    for (int nt = 0; nt < 8; ++nt) {
        const int col = nB * 128 + nt * 16 + lr;
        const float bv = bias ? bias[col] : 0.0f;
#pragma unroll
        for (int vr = 0; vr < 8; ++vr) {
            float v = acc[nt][vr] + bv;
            if (act == 1)      v = fmaxf(v, 0.0f);
            else if (act == 2) v = 0.5f * v * (1.0f + erff(v * 0.70710678118654752f));
            float* cp = &C[(size_t)(rowBase + vr) * N + col];
            if (addFlag) *cp = *cp + v; else *cp = v;
        }
    }
}

// ---------------- small utility kernels ----------------
__global__ void f32_to_f16_kernel(const float* __restrict__ s, _Float16* __restrict__ d, int n)
{
    int i = blockIdx.x * 256 + threadIdx.x;
    if (i < n) d[i] = (_Float16)s[i];
}

__global__ void copy_f32_kernel(const float* __restrict__ s, float* __restrict__ d, int n)
{
    int i = blockIdx.x * 256 + threadIdx.x;
    if (i < n) d[i] = s[i];
}

// ---------------- RWSE ----------------
__global__ void rw_count_kernel(const int* __restrict__ gsrc, const int* __restrict__ gdst,
                                float* __restrict__ A)
{
    int e = blockIdx.x * 256 + threadIdx.x;
    if (e >= E_GLOB) return;
    int s = gsrc[e], d = gdst[e];
    int gg = s >> 7;
    atomicAdd(&A[(size_t)gg * NPG * NPG + (size_t)(s & 127) * NPG + (d & 127)], 1.0f);
}

__global__ void __launch_bounds__(128)
rw_norm_kernel(float* T)   // row-normalize in place; 1 block per row
{
    __shared__ float red[128];
    int row = blockIdx.x;          // g*128 + i
    int t = threadIdx.x;
    float v = T[(size_t)row * NPG + t];
    red[t] = v;
    __syncthreads();
    for (int s = 64; s > 0; s >>= 1) {
        if (t < s) red[t] += red[t + s];
        __syncthreads();
    }
    float denom = fmaxf(red[0], 1.0f);
    T[(size_t)row * NPG + t] = v / denom;
}

__global__ void rw_diag_kernel(const float* __restrict__ P, float* __restrict__ rwse, int s)
{
    int i = blockIdx.x * 256 + threadIdx.x;     // 0..4095  (g*128 + r)
    if (i >= N_TOTAL) return;
    int gg = i >> 7, r = i & 127;
    rwse[(size_t)i * RW_STEPS + s] = P[(size_t)gg * NPG * NPG + (size_t)r * NPG + r];
}

__global__ void __launch_bounds__(128)
rw_matmul_kernel(const float* __restrict__ P, const float* __restrict__ T, float* __restrict__ Q)
{
    __shared__ float pr[128];
    int row = blockIdx.x;          // g*128 + i
    int gg = row >> 7;
    int t = threadIdx.x;
    pr[t] = P[(size_t)row * NPG + t];
    __syncthreads();
    const float* Tg = T + (size_t)gg * NPG * NPG;
    float acc = 0.0f;
#pragma unroll 8
    for (int k = 0; k < NPG; ++k) acc += pr[k] * Tg[(size_t)k * NPG + t];
    Q[(size_t)row * NPG + t] = acc;
}

// base[n] = atom_emb[x_ids[n]] + relu(rwse[n] @ rwse_W + rwse_b)
__global__ void __launch_bounds__(128)
base_kernel(const float* __restrict__ rwse, const float* __restrict__ rwse_W,
            const float* __restrict__ rwse_b, const float* __restrict__ atom_emb,
            const int* __restrict__ x_ids, float* __restrict__ base)
{
    __shared__ float rw[RW_STEPS];
    int n = blockIdx.x, t = threadIdx.x;
    if (t < RW_STEPS) rw[t] = rwse[(size_t)n * RW_STEPS + t];
    __syncthreads();
    float acc = rwse_b[t];
#pragma unroll
    for (int k = 0; k < RW_STEPS; ++k) acc += rw[k] * rwse_W[(size_t)k * H_DIM + t];
    acc = fmaxf(acc, 0.0f);
    base[(size_t)n * H_DIM + t] = atom_emb[(size_t)x_ids[n] * H_DIM + t] + acc;
}

__global__ void h0_kernel(const float* __restrict__ base, const int* __restrict__ node_ids,
                          float* __restrict__ h)
{
    int tid = blockIdx.x * 256 + threadIdx.x;       // FLAT_N * 32
    if (tid >= FLAT_N * 32) return;
    int f = tid >> 5;
    int c = (tid & 31) * 4;
    int nv = node_ids[f];
    float* hp = h + (size_t)f * H_DIM + c;
    if (nv >= 0) {
        int nc = nv < N_TOTAL ? nv : N_TOTAL - 1;
        const float* bp = base + (size_t)nc * H_DIM + c;
        hp[0] = bp[0]; hp[1] = bp[1]; hp[2] = bp[2]; hp[3] = bp[3];
    } else {
        hp[0] = 0.f; hp[1] = 0.f; hp[2] = 0.f; hp[3] = 0.f;
    }
}

// ---------------- GNN segment kernels ----------------
__global__ void intra_agg_kernel(const float* __restrict__ h, const int* __restrict__ isrc,
                                 const int* __restrict__ idst, const int* __restrict__ ea_ids,
                                 const float* __restrict__ bond, float* __restrict__ z)
{
    int tid = blockIdx.x * 256 + threadIdx.x;       // E_INTRA * 32
    if (tid >= E_INTRA * 32) return;
    int e = tid >> 5;
    int c = (tid & 31) * 4;
    int s = isrc[e], d = idst[e];
    const float* hp = h + (size_t)s * H_DIM + c;
    const float* bp = bond + (size_t)(ea_ids[e] - 1) * H_DIM + c;
    float* zp = z + (size_t)d * H_DIM + c;
#pragma unroll
    for (int j = 0; j < 4; ++j) atomicAdd(&zp[j], fmaxf(hp[j] + bp[j], 0.0f));
}

__global__ void gpool_kernel(const float* __restrict__ h, const int* __restrict__ node_ids,
                             float* __restrict__ gpool)
{
    int tid = blockIdx.x * 256 + threadIdx.x;       // FLAT_N * 32
    if (tid >= FLAT_N * 32) return;
    int f = tid >> 5;
    int c = (tid & 31) * 4;
    int nv = node_ids[f];
    if (nv < 0) return;                              // valid mask
    int nc = nv < N_TOTAL ? nv : N_TOTAL - 1;
    const float* hp = h + (size_t)f * H_DIM + c;
    float* gp = gpool + (size_t)nc * H_DIM + c;
#pragma unroll
    for (int j = 0; j < 4; ++j) atomicAdd(&gp[j], hp[j]);
}

__global__ void gagg_kernel(const float* __restrict__ gpool, const int* __restrict__ gsrc,
                            const int* __restrict__ gdst, const int* __restrict__ ea_ids,
                            const float* __restrict__ bond, float* __restrict__ gagg)
{
    int tid = blockIdx.x * 256 + threadIdx.x;       // E_GLOB * 32
    if (tid >= E_GLOB * 32) return;
    int e = tid >> 5;
    int c = (tid & 31) * 4;
    const float* sp = gpool + (size_t)gsrc[e] * H_DIM + c;
    const float* bp = bond + (size_t)(ea_ids[e] - 1) * H_DIM + c;
    float* dp = gagg + (size_t)gdst[e] * H_DIM + c;
#pragma unroll
    for (int j = 0; j < 4; ++j) atomicAdd(&dp[j], fmaxf(sp[j] + bp[j], 0.0f));
}

// z (already holds agg) += h + gagg[nid]
__global__ void zfin_kernel(const float* __restrict__ h, const float* __restrict__ gagg,
                            const int* __restrict__ node_ids, float* __restrict__ z)
{
    int tid = blockIdx.x * 256 + threadIdx.x;       // FLAT_N * 32
    if (tid >= FLAT_N * 32) return;
    int f = tid >> 5;
    int c = (tid & 31) * 4;
    int nv = node_ids[f];
    int nc = nv < 0 ? 0 : (nv < N_TOTAL ? nv : N_TOTAL - 1);
    const float* hp = h + (size_t)f * H_DIM + c;
    const float* gp = gagg + (size_t)nc * H_DIM + c;
    float* zp = z + (size_t)f * H_DIM + c;
#pragma unroll
    for (int j = 0; j < 4; ++j) zp[j] += hp[j] + gp[j];
}

__global__ void mask_kernel(const int* __restrict__ node_ids, float* __restrict__ h)
{
    int f = blockIdx.x * 256 + threadIdx.x;
    if (f >= FLAT_N) return;
    if (node_ids[f] < 0) {
        float* p = h + (size_t)f * H_DIM;
        for (int c = 0; c < H_DIM; ++c) p[c] = 0.0f;
    }
}

// ---------------- readout kernels ----------------
__global__ void tok_gather_kernel(const float* __restrict__ h, float* __restrict__ xt)
{
    int tid = blockIdx.x * 256 + threadIdx.x;       // S_SUB * 32
    if (tid >= S_SUB * 32) return;
    int s = tid >> 5;
    int c = (tid & 31) * 4;
    const float* hp = h + (size_t)s * K_SUB * H_DIM + c;    // row s*16 of h
    float* xp = xt + (size_t)s * H_DIM + c;
    xp[0] = hp[0]; xp[1] = hp[1]; xp[2] = hp[2]; xp[3] = hp[3];
}

__global__ void lpw_kernel(const float* __restrict__ lp, float* __restrict__ lpw)
{
    int i = blockIdx.x * 256 + threadIdx.x;
    if (i >= S_SUB) return;
    float v = lp[i];
    lpw[i] = __builtin_isfinite(v) ? v : 0.0f;
}

__global__ void __launch_bounds__(128)
ln_kernel(const float* __restrict__ X, float* __restrict__ Y,
          const float* __restrict__ g, const float* __restrict__ b)
{
    __shared__ float r1[128], r2[128];
    int row = blockIdx.x, t = threadIdx.x;
    float x = X[(size_t)row * H_DIM + t];
    r1[t] = x; r2[t] = x * x;
    __syncthreads();
    for (int s = 64; s > 0; s >>= 1) {
        if (t < s) { r1[t] += r1[t + s]; r2[t] += r2[t + s]; }
        __syncthreads();
    }
    float mu = r1[0] * (1.0f / H_DIM);
    float var = r2[0] * (1.0f / H_DIM) - mu * mu;
    Y[(size_t)row * H_DIM + t] = (x - mu) * rsqrtf(var + 1e-5f) * g[t] + b[t];
}

// one thread per (n, head, i): softmax over the M_SUB=4 keys
__global__ void attn_kernel(const float* __restrict__ qkv, const float* __restrict__ lpw,
                            const float* __restrict__ alphap, float* __restrict__ o)
{
    int t = blockIdx.x * 256 + threadIdx.x;         // N_TOTAL * NHEAD * M_SUB
    if (t >= N_TOTAL * NHEAD * M_SUB) return;
    int n = t >> 4;
    int hh = (t >> 2) & 3;
    int i = t & 3;
    const float alpha = alphap[0];
    const float scale = 0.17677669529663687f;       // 1/sqrt(32)

    float q[DHEAD];
    const float* qp = qkv + (size_t)(n * 4 + i) * 384 + hh * DHEAD;
#pragma unroll
    for (int d = 0; d < DHEAD; ++d) q[d] = qp[d];

    float sc[M_SUB];
#pragma unroll
    for (int j = 0; j < M_SUB; ++j) {
        const float* kp = qkv + (size_t)(n * 4 + j) * 384 + 128 + hh * DHEAD;
        float a = 0.0f;
#pragma unroll
        for (int d = 0; d < DHEAD; ++d) a += q[d] * kp[d];
        sc[j] = a * scale + alpha * lpw[n * 4 + j];
    }
    float mx = fmaxf(fmaxf(sc[0], sc[1]), fmaxf(sc[2], sc[3]));
    float ex[M_SUB], sum = 0.0f;
#pragma unroll
    for (int j = 0; j < M_SUB; ++j) { ex[j] = expf(sc[j] - mx); sum += ex[j]; }
    float inv = 1.0f / sum;

    float out[DHEAD];
#pragma unroll
    for (int d = 0; d < DHEAD; ++d) out[d] = 0.0f;
#pragma unroll
    for (int j = 0; j < M_SUB; ++j) {
        const float* vp = qkv + (size_t)(n * 4 + j) * 384 + 256 + hh * DHEAD;
        float pj = ex[j] * inv;
#pragma unroll
        for (int d = 0; d < DHEAD; ++d) out[d] += pj * vp[d];
    }
    float* op = o + (size_t)(n * 4 + i) * H_DIM + hh * DHEAD;
#pragma unroll
    for (int d = 0; d < DHEAD; ++d) op[d] = out[d];
}

// node_emb[n] = LN(mean over M tokens of xt[n*4 + i])
__global__ void __launch_bounds__(128)
meanln_kernel(const float* __restrict__ xt, float* __restrict__ nemb,
              const float* __restrict__ g, const float* __restrict__ b)
{
    __shared__ float r1[128], r2[128];
    int n = blockIdx.x, t = threadIdx.x;
    float x = 0.25f * (xt[(size_t)(n * 4 + 0) * H_DIM + t] +
                       xt[(size_t)(n * 4 + 1) * H_DIM + t] +
                       xt[(size_t)(n * 4 + 2) * H_DIM + t] +
                       xt[(size_t)(n * 4 + 3) * H_DIM + t]);
    r1[t] = x; r2[t] = x * x;
    __syncthreads();
    for (int s = 64; s > 0; s >>= 1) {
        if (t < s) { r1[t] += r1[t + s]; r2[t] += r2[t + s]; }
        __syncthreads();
    }
    float mu = r1[0] * (1.0f / H_DIM);
    float var = r2[0] * (1.0f / H_DIM) - mu * mu;
    nemb[(size_t)n * H_DIM + t] = (x - mu) * rsqrtf(var + 1e-5f) * g[t] + b[t];
}

__global__ void pool_kernel(const float* __restrict__ nemb, const int* __restrict__ batch,
                            float* __restrict__ out)
{
    int tid = blockIdx.x * 256 + threadIdx.x;       // N_TOTAL * 32
    if (tid >= N_TOTAL * 32) return;
    int n = tid >> 5;
    int c = (tid & 31) * 4;
    int bb = batch[n];
    const float* np = nemb + (size_t)n * H_DIM + c;
    float* op = out + (size_t)bb * H_DIM + c;
#pragma unroll
    for (int j = 0; j < 4; ++j) atomicAdd(&op[j], np[j]);
}

// =====================================================================
// launcher
// =====================================================================
static inline void launch_gemm(const float* A, const _Float16* Bw, const float* bias,
                               float* C, int M, int N, int K, int act, int add,
                               hipStream_t stream)
{
    dim3 grid(M / 128, N / 128);
    gemm_wmma<<<grid, 256, 0, stream>>>(A, Bw, bias, C, M, N, K, act, add);
}

extern "C" void kernel_launch(void* const* d_in, const int* in_sizes, int n_in,
                              void* d_out, int out_size, void* d_ws, size_t ws_size,
                              hipStream_t stream)
{
    (void)in_sizes; (void)n_in; (void)out_size; (void)ws_size;

    const int*   edge_index   = (const int*)d_in[0];
    const int*   gsrc         = edge_index;
    const int*   gdst         = edge_index + E_GLOB;
    const int*   intra_ei     = (const int*)d_in[1];
    const int*   isrc         = intra_ei;
    const int*   idst         = intra_ei + E_INTRA;
    const int*   ea_glob_ids  = (const int*)d_in[2];
    const int*   intra_ea_ids = (const int*)d_in[3];
    const int*   node_ids     = (const int*)d_in[4];
    const int*   x_ids        = (const int*)d_in[5];
    const int*   batch        = (const int*)d_in[6];
    const float* log_probs    = (const float*)d_in[7];
    const float* atom_emb     = (const float*)d_in[8];
    const float* bond_emb     = (const float*)d_in[9];
    const float* rwse_W       = (const float*)d_in[10];
    const float* rwse_b       = (const float*)d_in[11];
    const float* gnn_W1       = (const float*)d_in[12];
    const float* gnn_b1       = (const float*)d_in[13];
    const float* gnn_W2       = (const float*)d_in[14];
    const float* gnn_b2       = (const float*)d_in[15];
    const float* ro_ln1_g     = (const float*)d_in[16];
    const float* ro_ln1_b     = (const float*)d_in[17];
    const float* ro_Wqkv      = (const float*)d_in[18];
    const float* ro_bqkv      = (const float*)d_in[19];
    const float* ro_Wo        = (const float*)d_in[20];
    const float* ro_bo        = (const float*)d_in[21];
    const float* ro_ln2_g     = (const float*)d_in[22];
    const float* ro_ln2_b     = (const float*)d_in[23];
    const float* ro_Wf1       = (const float*)d_in[24];
    const float* ro_bf1       = (const float*)d_in[25];
    const float* ro_Wf2       = (const float*)d_in[26];
    const float* ro_bf2       = (const float*)d_in[27];
    const float* out_ln_g     = (const float*)d_in[28];
    const float* out_ln_b     = (const float*)d_in[29];
    const float* ht_alpha     = (const float*)d_in[30];

    // ---------------- workspace layout ----------------
    const size_t MB = 1024ull * 1024ull;
    const size_t H_BYTES = (size_t)FLAT_N * H_DIM * 4;      // 134,217,728
    uint8_t* ws = (uint8_t*)d_ws;
    float* h  = (float*)(ws);                                // region 0
    uint8_t* Breg = ws + H_BYTES;                            // region B (134MB, reused)
    uint8_t* Creg = ws + 2 * H_BYTES;                        // region C (small)

    // region B, RWSE phase
    float* Tm   = (float*)(Breg + 0);                        // 2MB
    float* P    = (float*)(Breg + 2 * MB);
    float* P2   = (float*)(Breg + 4 * MB);
    float* rwse = (float*)(Breg + 6 * MB);                   // 256KB
    float* base = (float*)(Breg + 8 * MB);                   // 2MB
    // region B, GNN phase (z doubles as agg accumulator and as GEMM1 output)
    float* z    = (float*)(Breg);                            // 134MB
    // region B, readout phase
    float* xt   = (float*)(Breg + 0);                        // 8MB
    float* xn   = (float*)(Breg + 8 * MB);                   // 8MB
    float* qkv  = (float*)(Breg + 16 * MB);                  // 24MB
    float* oatt = (float*)(Breg + 48 * MB);                  // 8MB
    float* f1   = (float*)(Breg + 56 * MB);                  // 32MB
    float* nemb = (float*)(Breg + 92 * MB);                  // 2MB
    float* lpw  = (float*)(Breg + 96 * MB);                  // 64KB

    // region C
    float* gpool = (float*)(Creg);                           // 2MB
    float* gagg  = (float*)(Creg + 2 * MB);                  // 2MB
    _Float16* w16    = (_Float16*)(Creg + 4 * MB);           // 1MB of f16 weights
    _Float16* w1_16   = w16;                                 // 4*128*128
    _Float16* w2_16   = w16 + 65536;                         // 4*128*128
    _Float16* wqkv16  = w16 + 131072;                        // 2*128*384
    _Float16* wo16    = w16 + 229376;                        // 2*128*128
    _Float16* wf1_16  = w16 + 262144;                        // 2*128*512
    _Float16* wf2_16  = w16 + 393216;                        // 2*512*128

    // ---------------- convert weights to f16 ----------------
    f32_to_f16_kernel<<<(65536 + 255) / 256, 256, 0, stream>>>(gnn_W1, w1_16, 65536);
    f32_to_f16_kernel<<<(65536 + 255) / 256, 256, 0, stream>>>(gnn_W2, w2_16, 65536);
    f32_to_f16_kernel<<<(98304 + 255) / 256, 256, 0, stream>>>(ro_Wqkv, wqkv16, 98304);
    f32_to_f16_kernel<<<(32768 + 255) / 256, 256, 0, stream>>>(ro_Wo, wo16, 32768);
    f32_to_f16_kernel<<<(131072 + 255) / 256, 256, 0, stream>>>(ro_Wf1, wf1_16, 131072);
    f32_to_f16_kernel<<<(131072 + 255) / 256, 256, 0, stream>>>(ro_Wf2, wf2_16, 131072);

    // ---------------- RWSE: random-walk diagonals ----------------
    (void)hipMemsetAsync(Tm, 0, (size_t)B_GRAPHS * NPG * NPG * 4, stream);
    rw_count_kernel<<<(E_GLOB + 255) / 256, 256, 0, stream>>>(gsrc, gdst, Tm);
    rw_norm_kernel<<<B_GRAPHS * NPG, 128, 0, stream>>>(Tm);
    copy_f32_kernel<<<(B_GRAPHS * NPG * NPG + 255) / 256, 256, 0, stream>>>(
        Tm, P, B_GRAPHS * NPG * NPG);
    for (int s = 0; s < RW_STEPS; ++s) {
        rw_diag_kernel<<<(N_TOTAL + 255) / 256, 256, 0, stream>>>(P, rwse, s);
        if (s < RW_STEPS - 1) {
            rw_matmul_kernel<<<B_GRAPHS * NPG, 128, 0, stream>>>(P, Tm, P2);
            float* tmp = P; P = P2; P2 = tmp;
        }
    }
    base_kernel<<<N_TOTAL, 128, 0, stream>>>(rwse, rwse_W, rwse_b, atom_emb, x_ids, base);
    h0_kernel<<<(FLAT_N * 32) / 256, 256, 0, stream>>>(base, node_ids, h);

    // ---------------- GNN layers ----------------
    for (int l = 0; l < NL_GNN; ++l) {
        (void)hipMemsetAsync(z, 0, H_BYTES, stream);
        intra_agg_kernel<<<(E_INTRA * 32) / 256, 256, 0, stream>>>(
            h, isrc, idst, intra_ea_ids, bond_emb, z);
        (void)hipMemsetAsync(gpool, 0, (size_t)N_TOTAL * H_DIM * 4, stream);
        gpool_kernel<<<(FLAT_N * 32) / 256, 256, 0, stream>>>(h, node_ids, gpool);
        (void)hipMemsetAsync(gagg, 0, (size_t)N_TOTAL * H_DIM * 4, stream);
        gagg_kernel<<<(E_GLOB * 32) / 256, 256, 0, stream>>>(
            gpool, gsrc, gdst, ea_glob_ids, bond_emb, gagg);
        zfin_kernel<<<(FLAT_N * 32) / 256, 256, 0, stream>>>(h, gagg, node_ids, z);
        // t = relu(z @ W1 + b1)   (in-place over z — block-disjoint rows)
        launch_gemm(z, w1_16 + l * 16384, gnn_b1 + l * H_DIM, z,
                    FLAT_N, H_DIM, H_DIM, /*relu*/1, /*write*/0, stream);
        // h += t @ W2 + b2
        launch_gemm(z, w2_16 + l * 16384, gnn_b2 + l * H_DIM, h,
                    FLAT_N, H_DIM, H_DIM, 0, /*add*/1, stream);
        mask_kernel<<<(FLAT_N + 255) / 256, 256, 0, stream>>>(node_ids, h);
    }

    // ---------------- readout transformer ----------------
    tok_gather_kernel<<<(S_SUB * 32) / 256, 256, 0, stream>>>(h, xt);
    lpw_kernel<<<(S_SUB + 255) / 256, 256, 0, stream>>>(log_probs, lpw);
    for (int l = 0; l < NL_RO; ++l) {
        ln_kernel<<<S_SUB, 128, 0, stream>>>(xt, xn, ro_ln1_g + l * H_DIM, ro_ln1_b + l * H_DIM);
        launch_gemm(xn, wqkv16 + l * 49152, ro_bqkv + l * 384, qkv,
                    S_SUB, 384, H_DIM, 0, 0, stream);
        attn_kernel<<<(N_TOTAL * NHEAD * M_SUB) / 256, 256, 0, stream>>>(qkv, lpw, ht_alpha, oatt);
        launch_gemm(oatt, wo16 + l * 16384, ro_bo + l * H_DIM, xt,
                    S_SUB, H_DIM, H_DIM, 0, /*add*/1, stream);
        ln_kernel<<<S_SUB, 128, 0, stream>>>(xt, xn, ro_ln2_g + l * H_DIM, ro_ln2_b + l * H_DIM);
        launch_gemm(xn, wf1_16 + l * 65536, ro_bf1 + l * FFN_DIM, f1,
                    S_SUB, FFN_DIM, H_DIM, /*gelu*/2, 0, stream);
        launch_gemm(f1, wf2_16 + l * 65536, ro_bf2 + l * H_DIM, xt,
                    S_SUB, H_DIM, FFN_DIM, 0, /*add*/1, stream);
    }

    // ---------------- output ----------------
    meanln_kernel<<<N_TOTAL, 128, 0, stream>>>(xt, nemb, out_ln_g, out_ln_b);
    (void)hipMemsetAsync(d_out, 0, (size_t)B_GRAPHS * H_DIM * 4, stream);
    pool_kernel<<<(N_TOTAL * 32) / 256, 256, 0, stream>>>(nemb, batch, (float*)d_out);
}
